// MultiHeadSelfAttention_85942295593076
// MI455X (gfx1250) — matrix operand order repaired and checked by reading, not compiled, not run on hardware
//
#include <hip/hip_runtime.h>
#include <hip/hip_bf16.h>

typedef _Float16 f16;
typedef __attribute__((ext_vector_type(8)))  _Float16 v8h;
typedef __attribute__((ext_vector_type(16))) _Float16 v16h;
typedef __attribute__((ext_vector_type(8)))  float    v8f;

#define DMODEL 1024
#define NHEAD  16
#define HDIM   64
#define SEQ    2048
#define BATCH  2
#define NTOK   (BATCH*SEQ)   // 4096
#define BHCNT  (BATCH*NHEAD) // 32

static __device__ __forceinline__ v16h join8(v8h lo, v8h hi){
  v16h r;
#pragma unroll
  for (int i=0;i<8;++i){ r[i]=lo[i]; r[i+8]=hi[i]; }
  return r;
}

// A fragment: 16x32 (MxK) from row-major src. Lane m = lane%16, half = lane/16.
// elements 0..7 -> K = half*8 + i ; elements 8..15 -> K = 16 + half*8 + (i-8)
static __device__ __forceinline__ v16h load_a_frag(const f16* __restrict__ src, int ld, int row0, int k0){
  const int lane = threadIdx.x & 31;
  const int m = lane & 15, half = lane >> 4;
  const f16* p = src + (size_t)(row0 + m) * (size_t)ld + k0 + half*8;
  v8h lo = *(const v8h*)p;
  v8h hi = *(const v8h*)(p + 16);
  return join8(lo, hi);
}

// B fragment: 32x16 (KxN) from Bt stored as [n][k] row-major.
// Lane n = lane%16, half = lane/16; elements 0..7 -> K = half*8 + i ; elements 8..15 -> K = 16 + half*8 + (i-8)
static __device__ __forceinline__ v16h load_b_frag(const f16* __restrict__ bt, int ld, int k0, int n0){
  const int lane = threadIdx.x & 31;
  const int n = lane & 15, half = lane >> 4;
  const f16* p = bt + (size_t)(n0 + n) * (size_t)ld + k0 + half*8;
  v8h lo = *(const v8h*)p;
  v8h hi = *(const v8h*)(p + 16);
  return join8(lo, hi);
}

static __device__ __forceinline__ v8f wmma_f16(v16h a, v16h b, v8f c){
  return __builtin_amdgcn_wmma_f32_16x16x32_f16(false, a, false, b, (short)0, c, false, false);
}

// ---------------- conversion / transpose kernels ----------------

__global__ void k_cvt_f16(const float* __restrict__ in, f16* __restrict__ out, int n){
  int i = blockIdx.x * blockDim.x + threadIdx.x;
  if (i < n) out[i] = (f16)in[i];
}

// out[c][r] = in[r][c] ; in is rows_in x cols_in fp32, out is cols_in x rows_in f16
__global__ void k_transpose_f16(const float* __restrict__ in, f16* __restrict__ out,
                                int rows_in, int cols_in){
  int i = blockIdx.x * blockDim.x + threadIdx.x;
  int total = rows_in * cols_in;
  if (i >= total) return;
  int c = i / rows_in;
  int r = i - c * rows_in;
  out[(size_t)c * rows_in + r] = (f16)in[(size_t)r * cols_in + c];
}

// -------- shared GEMM core: 32x64 tile per wave = acc[2][4], K fully unrolled

static __device__ __forceinline__ void gemm_core(const f16* __restrict__ A,
                                                 const f16* __restrict__ Bt,
                                                 int row0, int col0,
                                                 v8f acc[2][4]){
  const int K = DMODEL;  // compile-time so the loop fully unrolls
#pragma unroll
  for (int k0 = 0; k0 < K; k0 += 32){
    if (k0 + 128 < K){
      __builtin_prefetch(A  + (size_t)(row0 + (threadIdx.x & 15)) * (size_t)K + k0 + 128, 0, 3);
      __builtin_prefetch(Bt + (size_t)(col0 + (threadIdx.x & 15)) * (size_t)K + k0 + 128, 0, 3);
    }
    // stage all fragments first so load clauses can pipeline across the WMMAs
    v16h a0 = load_a_frag(A, K, row0,      k0);
    v16h a1 = load_a_frag(A, K, row0 + 16, k0);
    v16h b[4];
#pragma unroll
    for (int j = 0; j < 4; ++j) b[j] = load_b_frag(Bt, K, k0, col0 + j*16);
#pragma unroll
    for (int j = 0; j < 4; ++j){
      acc[0][j] = wmma_f16(a0, b[j], acc[0][j]);
      acc[1][j] = wmma_f16(a1, b[j], acc[1][j]);
    }
  }
}

// ---------------- QKV projection: qkv = xh @ Wqkv + b ----------------
// Writes q (scaled by 1/8), k row-major [bh][s][hd]; v transposed [bh][hd][s].

__global__ __launch_bounds__(256) void k_qkv_gemm(const f16* __restrict__ xh,
                                                  const f16* __restrict__ wt,
                                                  const float* __restrict__ bias,
                                                  f16* __restrict__ q,
                                                  f16* __restrict__ kk,
                                                  f16* __restrict__ vt){
  const int wid  = threadIdx.x >> 5;
  const int row0 = blockIdx.y * 64  + (wid >> 2) * 32;
  const int col0 = blockIdx.x * 256 + (wid & 3)  * 64;
  v8f acc[2][4] = {};
  gemm_core(xh, wt, row0, col0, acc);

  const int lane = threadIdx.x & 31;
  const int g = lane >> 4, n = lane & 15;
#pragma unroll
  for (int mi = 0; mi < 2; ++mi){
#pragma unroll
    for (int ni = 0; ni < 4; ++ni){
#pragma unroll
      for (int r = 0; r < 8; ++r){
        int gm = row0 + mi*16 + 8*g + r;          // token row 0..4095
        int gn = col0 + ni*16 + n;                // qkv col 0..3071
        float val = acc[mi][ni][r] + bias[gn];
        int b = gm >> 11, s = gm & (SEQ - 1);
        int which = gn >> 10, c = gn & (DMODEL - 1);
        int h = c >> 6, hd = c & 63;
        size_t bh = (size_t)(b * NHEAD + h);
        if (which == 0)      q [(bh*SEQ + s)*HDIM + hd] = (f16)(val * 0.125f);
        else if (which == 1) kk[(bh*SEQ + s)*HDIM + hd] = (f16)val;
        else                 vt[(bh*HDIM + hd)*SEQ + s] = (f16)val;
      }
    }
  }
}

// ---------------- flash attention: one wave per 32 query rows ----------------

__global__ __launch_bounds__(128) void k_attn(const f16* __restrict__ q,
                                              const f16* __restrict__ kmat,
                                              const f16* __restrict__ vt,
                                              f16* __restrict__ attn){
  __shared__ __attribute__((aligned(16))) f16 pbuf[4][32][32];   // 8 KB
  const int wid  = threadIdx.x >> 5;
  const int lane = threadIdx.x & 31;
  const int g = lane >> 4, n = lane & 15;
  const int bh   = blockIdx.y;
  const int row0 = blockIdx.x * 128 + wid * 32;

  const f16* qb = q    + (size_t)bh * SEQ * HDIM;
  const f16* kb = kmat + (size_t)bh * SEQ * HDIM;
  const f16* vb = vt   + (size_t)bh * HDIM * SEQ;

  // Q fragments are loop-invariant (already scaled by 1/sqrt(HD))
  v16h aq[2][2];
#pragma unroll
  for (int mi = 0; mi < 2; ++mi){
    aq[mi][0] = load_a_frag(qb, HDIM, row0 + mi*16, 0);
    aq[mi][1] = load_a_frag(qb, HDIM, row0 + mi*16, 32);
  }

  v8f acc[2][4] = {};
  float mrun[2][8], lrun[2][8];
#pragma unroll
  for (int mi = 0; mi < 2; ++mi)
#pragma unroll
    for (int r = 0; r < 8; ++r){ mrun[mi][r] = -3.0e38f; lrun[mi][r] = 0.f; }

  for (int t0 = 0; t0 < SEQ; t0 += 32){
    // scores: 32 rows x 32 keys, reduction over HD=64 (two K=32 chunks)
    v8f s[2][2] = {};
    {
      v16h bk00 = load_b_frag(kb, HDIM, 0,  t0);
      v16h bk01 = load_b_frag(kb, HDIM, 32, t0);
      v16h bk10 = load_b_frag(kb, HDIM, 0,  t0 + 16);
      v16h bk11 = load_b_frag(kb, HDIM, 32, t0 + 16);
#pragma unroll
      for (int mi = 0; mi < 2; ++mi){
        s[mi][0] = wmma_f16(aq[mi][0], bk00, s[mi][0]);
        s[mi][0] = wmma_f16(aq[mi][1], bk01, s[mi][0]);
        s[mi][1] = wmma_f16(aq[mi][0], bk10, s[mi][1]);
        s[mi][1] = wmma_f16(aq[mi][1], bk11, s[mi][1]);
      }
    }
    // online softmax; C layout: VGPR r holds row 8*g+r, col n -> lane-local
#pragma unroll
    for (int mi = 0; mi < 2; ++mi){
#pragma unroll
      for (int r = 0; r < 8; ++r){
        float mx = fmaxf(s[mi][0][r], s[mi][1][r]);
#pragma unroll
        for (int msk = 1; msk < 16; msk <<= 1) mx = fmaxf(mx, __shfl_xor(mx, msk, 32));
        float nm = fmaxf(mrun[mi][r], mx);
        float p0 = __expf(s[mi][0][r] - nm);
        float p1 = __expf(s[mi][1][r] - nm);
        float sm = p0 + p1;
#pragma unroll
        for (int msk = 1; msk < 16; msk <<= 1) sm += __shfl_xor(sm, msk, 32);
        float corr = __expf(mrun[mi][r] - nm);
        lrun[mi][r] = lrun[mi][r] * corr + sm;
        mrun[mi][r] = nm;
#pragma unroll
        for (int j = 0; j < 4; ++j) acc[mi][j][r] *= corr;
        // bounce P to LDS (row-major 32x32) to re-layout C-tile -> A-fragment
        pbuf[wid][mi*16 + 8*g + r][n]      = (f16)p0;
        pbuf[wid][mi*16 + 8*g + r][16 + n] = (f16)p1;
      }
    }
    // P(32x32) @ V(32x64): LDS ops are in-order within a wave, no barrier needed
    {
      const int m = lane & 15;
      v16h ap[2];
#pragma unroll
      for (int mi = 0; mi < 2; ++mi){
        const f16* pp = &pbuf[wid][mi*16 + m][g * 8];
        v8h lo = *(const v8h*)pp;
        v8h hi = *(const v8h*)(pp + 16);
        ap[mi] = join8(lo, hi);
      }
#pragma unroll
      for (int j = 0; j < 4; ++j){
        v16h bv = load_b_frag(vb, SEQ, t0, j * 16);
        acc[0][j] = wmma_f16(ap[0], bv, acc[0][j]);
        acc[1][j] = wmma_f16(ap[1], bv, acc[1][j]);
      }
    }
  }

  // normalize and store to attn_h [b*SEQ+s][h*64+hd] (f16 for the out-proj GEMM)
  const int b = bh >> 4, h = bh & 15;
#pragma unroll
  for (int mi = 0; mi < 2; ++mi){
#pragma unroll
    for (int j = 0; j < 4; ++j){
#pragma unroll
      for (int r = 0; r < 8; ++r){
        int s = row0 + mi*16 + 8*g + r;
        float o = acc[mi][j][r] / lrun[mi][r];
        attn[((size_t)(b * SEQ + s)) * DMODEL + h * HDIM + j * 16 + n] = (f16)o;
      }
    }
  }
}

// ---------------- output projection: out = attn_h @ Wout + b_out (fp32) ------

__global__ __launch_bounds__(256) void k_out_gemm(const f16* __restrict__ ah,
                                                   const f16* __restrict__ wt,
                                                   const float* __restrict__ bias,
                                                   float* __restrict__ out){
  const int wid  = threadIdx.x >> 5;
  const int row0 = blockIdx.y * 64  + (wid >> 2) * 32;
  const int col0 = blockIdx.x * 256 + (wid & 3)  * 64;
  v8f acc[2][4] = {};
  gemm_core(ah, wt, row0, col0, acc);

  const int lane = threadIdx.x & 31;
  const int g = lane >> 4, n = lane & 15;
#pragma unroll
  for (int mi = 0; mi < 2; ++mi){
#pragma unroll
    for (int ni = 0; ni < 4; ++ni){
#pragma unroll
      for (int r = 0; r < 8; ++r){
        int gm = row0 + mi*16 + 8*g + r;
        int gn = col0 + ni*16 + n;
        out[(size_t)gm * DMODEL + gn] = acc[mi][ni][r] + bias[gn];
      }
    }
  }
}

// ---------------- host launch ----------------

extern "C" void kernel_launch(void* const* d_in, const int* in_sizes, int n_in,
                              void* d_out, int out_size, void* d_ws, size_t ws_size,
                              hipStream_t stream){
  (void)in_sizes; (void)n_in; (void)out_size; (void)ws_size;
  const float* x    = (const float*)d_in[0];
  const float* Wqkv = (const float*)d_in[1];
  const float* bqkv = (const float*)d_in[2];
  const float* Wout = (const float*)d_in[3];
  const float* bout = (const float*)d_in[4];
  float* out = (float*)d_out;

  char* ws = (char*)d_ws;
  size_t off = 0;
  auto alloc = [&](size_t bytes)->char*{
    char* p = ws + off; off += (bytes + 255) & ~(size_t)255; return p;
  };
  f16* xh  = (f16*)alloc((size_t)NTOK * DMODEL * 2);          // 8 MB
  f16* wqt = (f16*)alloc((size_t)3 * DMODEL * DMODEL * 2);    // 6 MB
  f16* wot = (f16*)alloc((size_t)DMODEL * DMODEL * 2);        // 2 MB
  f16* qb  = (f16*)alloc((size_t)BHCNT * SEQ * HDIM * 2);     // 8 MB
  f16* kb  = (f16*)alloc((size_t)BHCNT * SEQ * HDIM * 2);     // 8 MB
  f16* vtb = (f16*)alloc((size_t)BHCNT * HDIM * SEQ * 2);     // 8 MB
  f16* ah  = (f16*)alloc((size_t)NTOK * DMODEL * 2);          // 8 MB

  k_cvt_f16<<<(NTOK * DMODEL) / 256, 256, 0, stream>>>(x, xh, NTOK * DMODEL);
  k_transpose_f16<<<(3 * DMODEL * DMODEL) / 256, 256, 0, stream>>>(Wqkv, wqt, DMODEL, 3 * DMODEL);
  k_transpose_f16<<<(DMODEL * DMODEL) / 256, 256, 0, stream>>>(Wout, wot, DMODEL, DMODEL);

  k_qkv_gemm<<<dim3(3 * DMODEL / 256, NTOK / 64), 256, 0, stream>>>(xh, wqt, bqkv, qb, kb, vtb);
  k_attn   <<<dim3(SEQ / 128, BHCNT),             128, 0, stream>>>(qb, kb, vtb, ah);
  k_out_gemm<<<dim3(DMODEL / 256, NTOK / 64),     256, 0, stream>>>(ah, wot, bout, out);
}
